// NNLMDecoder_35373350650612
// MI455X (gfx1250) — matrix operand order repaired and checked by reading, not compiled, not run on hardware
//
#include <hip/hip_runtime.h>
#include <hip/hip_bf16.h>
#include <stdint.h>

typedef __attribute__((ext_vector_type(16))) __bf16 v16bf;
typedef __attribute__((ext_vector_type(8)))  float  v8f;
typedef __attribute__((ext_vector_type(4)))  unsigned int u32x4;
typedef __attribute__((ext_vector_type(8)))  int          i32x8;
typedef __attribute__((ext_vector_type(4)))  int          i32x4;

#if defined(__has_builtin)
#if __has_builtin(__builtin_amdgcn_tensor_load_to_lds) && __has_builtin(__builtin_amdgcn_s_wait_tensorcnt)
#define HAVE_TDM 1
#endif
#endif
#ifndef HAVE_TDM
#define HAVE_TDM 0
#endif

#define B_  64
#define T_  512
#define P_  8
#define V_  50000
#define O_  50000
#define E_  512
#define H_  2048
#define CS_ 5

static __device__ __forceinline__ unsigned short f2bf(float f) {
    unsigned int u = __float_as_uint(f);
    u += 0x8000u;                      // round-to-nearest
    return (unsigned short)(u >> 16);
}

// -------------------- zero d_out --------------------
__global__ void zero_out_k(float* p, int n) {
    int i = blockIdx.x * blockDim.x + threadIdx.x;
    if (i < n) p[i] = 0.0f;
}

// -------------------- n-gram features -> bonus in d_out --------------------
__global__ __launch_bounds__(512)
void feat_kernel(const int* __restrict__ pseq, const int* __restrict__ src,
                 const int* __restrict__ lens, float* __restrict__ out) {
    __shared__ int s[T_];
    __shared__ int kpos;
    const int b = blockIdx.x, t = threadIdx.x;
    s[t] = src[b * T_ + t];
    if (t == 0) kpos = 0x7FFFFFFF;
    __syncthreads();
    const int w1  = pseq[b * P_ + (P_ - 1)];
    const int w2a = pseq[b * P_ + (P_ - 2)];
    if (s[t] == w1) atomicMin(&kpos, t);
    __syncthreads();
    const int v = s[t];
    int big = 0, tri = 0, jp = 0x7FFFFFFF;
    for (int j = 0; j < T_; ++j) {
        int sj = s[j];
        if (j < T_ - 1 && sj == w1 && s[j + 1] == v) big = 1;
        if (j < T_ - 2 && sj == w2a && s[j + 1] == w1 && s[j + 2] == v) tri = 1;
        if (sj == v && j < jp) jp = j;
    }
    const int kp = kpos;
    const int inlist = (kp != 0x7FFFFFFF);
    const int reo = inlist && (jp < kp);
    if (t < lens[b]) {   // same token value -> same bonus, plain store is safe
        out[(size_t)b * O_ + v] = 0.5f * (float)(1 + big + tri + reo);
    }
}

// -------------------- context gather (fp32 -> bf16) --------------------
__global__ void gather_ctx_k(const int* __restrict__ pseq, const float* __restrict__ Et,
                             unsigned short* __restrict__ ctx) {
    int id = blockIdx.x * blockDim.x + threadIdx.x;
    if (id >= B_ * CS_ * E_) return;
    int b = id / (CS_ * E_);
    int r = id % (CS_ * E_);
    int j = r / E_;
    int e = r % E_;
    int tok = pseq[b * P_ + (P_ - CS_) + j];
    ctx[(size_t)b * (CS_ * E_) + r] = f2bf(Et[(size_t)tok * E_ + e]);
}

// -------------------- encoder_output -> hcat[:, 2048:] (bf16) --------------------
__global__ void enc_to_hcat_k(const float* __restrict__ enc, unsigned short* __restrict__ hcat) {
    int id = blockIdx.x * blockDim.x + threadIdx.x;
    if (id >= B_ * H_) return;
    int b = id / H_, e = id % H_;
    hcat[(size_t)b * (2 * H_) + H_ + e] = f2bf(enc[(size_t)b * H_ + e]);
}

// -------------------- TDM: load [64 x 32] bf16 tile (row stride lda) into LDS --------
// LDS rows padded 64B data + 16B pad (pad_interval=16DW, pad_amount=4DW) -> 80B rows,
// matching the As[64][20] fragment addressing.
static __device__ __forceinline__ void tdm_load_A(const unsigned short* gA,
                                                  unsigned ldsOff, int lda) {
#if HAVE_TDM
    unsigned long long ga = (unsigned long long)(uintptr_t)gA;
    u32x4 g0;
    g0.x = 1u;                                                 // count=1, user desc
    g0.y = ldsOff;                                             // lds_addr
    g0.z = (unsigned)(ga & 0xFFFFFFFFu);                       // global_addr[31:0]
    g0.w = (unsigned)((ga >> 32) & 0x01FFFFFFu) | (2u << 30);  // global_addr[56:32] | type=2
    i32x8 g1;
    g1[0] = (1 << 16) | (1 << 20) | (3 << 22) | (3 << 25);     // data_size=2B, pad_en, 16DW->+4DW
    g1[1] = (lda & 0xFFFF) << 16;                              // tensor_dim0[15:0]
    g1[2] = ((lda >> 16) & 0xFFFF) | (64 << 16);               // tensor_dim0[31:16] | tensor_dim1 lo
    g1[3] = (32 << 16);                                        // tensor_dim1 hi=0 | tile_dim0=32
    g1[4] = 64;                                                // tile_dim1=64, tile_dim2=0
    g1[5] = lda;                                               // tensor_dim0_stride[31:0]
    g1[6] = 0;                                                 // stride0 hi | dim1_stride lo
    g1[7] = 0;
    i32x4 z = {0, 0, 0, 0};
#if __clang_major__ >= 23
    i32x8 z8 = {0, 0, 0, 0, 0, 0, 0, 0};
    __builtin_amdgcn_tensor_load_to_lds(g0, g1, z, z, z8, 0);
#else
    __builtin_amdgcn_tensor_load_to_lds(g0, g1, z, z, 0);
#endif
#endif
}

// -------------------- WMMA GEMM: [64,K]bf16 @ [K,N]f32(->bf16) --------------------
// MODE 0: out_bf16 = tanh(acc + bias0[n])            (hidden layer)
// MODE 1: out_f32  = acc + bias0[n] + bias1[n] + out (output projection + bonus)
template <int MODE>
__global__ __launch_bounds__(512)
void gemm_wmma(const unsigned short* __restrict__ A, int lda,
               const float* __restrict__ B0, const float* __restrict__ B1, int Kb,
               int K, int N, int ldb,
               const float* __restrict__ bias0, const float* __restrict__ bias1,
               float* __restrict__ outF, unsigned short* __restrict__ outBF,
               int ldo) {
    __shared__ alignas(16) unsigned int   As[3][64][20];    // TDM triple buffer, 80B rows
    __shared__ alignas(16) unsigned short Bs[2][256][40];   // [n][k] bf16, 80B rows

    const int tid  = threadIdx.x;
    const int lane = tid & 31;
    const int wave = tid >> 5;
    const int wm   = wave & 3;        // m-tile (16 rows)
    const int wg   = wave >> 2;       // n-group (64 cols)
    const int n0blk = blockIdx.x * 256;
    const int hlf  = lane >> 4;
    const int l16  = lane & 15;
    const int nK   = K >> 5;

    // B staging coordinates (constant over k-loop)
    const int Bn = (tid & 63) << 2;     // local n (4 floats per thread per group)
    int ngc = n0blk + Bn;               // clamped global n (OOB cols computed, never stored)
    if (ngc > N - 4) ngc = N - 4;

    v8f acc[4] = {};
    float4 bReg[4];

    auto issue_B = [&](int kc) {
        const int k0 = kc << 5;
        const float* Bp = (k0 < Kb) ? (B0 + (size_t)k0 * ldb)
                                    : (B1 + (size_t)(k0 - Kb) * ldb);
#pragma unroll
        for (int g = 0; g < 4; ++g) {
            const int k = (tid >> 6) + g * 8;
            bReg[g] = *(const float4*)(Bp + (size_t)k * ldb + ngc);
        }
    };
    auto stage_A = [&](int kc, int bufA) {
#if HAVE_TDM
        if (wave == 0)
            tdm_load_A(A + (size_t)(kc << 5), (unsigned)(uintptr_t)&As[bufA][0][0], lda);
#else
        // fallback: plain copy, one b64 per thread
        const int Am = tid >> 3, Ac = (tid & 7) * 2, ldaU = lda >> 1;
        *(uint2*)&As[bufA][Am][Ac] =
            *(const uint2*)((const unsigned int*)A + (size_t)Am * ldaU + (kc << 4) + Ac);
#endif
    };

    stage_A(0, 0);
    issue_B(0);

    for (int kc = 0; kc < nK; ++kc) {
        const int bufB = kc & 1;
        const int bufA = kc % 3;
        // drain staged B regs -> LDS (fp32 -> bf16)
#pragma unroll
        for (int g = 0; g < 4; ++g) {
            const int k = (tid >> 6) + g * 8;
            const float4 v = bReg[g];
            Bs[bufB][Bn + 0][k] = f2bf(v.x);
            Bs[bufB][Bn + 1][k] = f2bf(v.y);
            Bs[bufB][Bn + 2][k] = f2bf(v.z);
            Bs[bufB][Bn + 3][k] = f2bf(v.w);
        }
        // software pipeline: kick off next chunk's streams before computing
        if (kc + 1 < nK) {
            issue_B(kc + 1);
            stage_A(kc + 1, (kc + 1) % 3);
        }
        // prefetch B two chunks ahead (32 rows x 1KB; 256 lines of 128B)
        if (kc + 2 < nK && tid < 256) {
            const int k0p = (kc + 2) << 5;
            const float* Bp2 = (k0p < Kb) ? (B0 + (size_t)k0p * ldb)
                                          : (B1 + (size_t)(k0p - Kb) * ldb);
            int pn = n0blk + ((tid & 7) << 5);
            if (pn > N - 32) pn = N - 32;
            __builtin_prefetch(Bp2 + (size_t)(tid >> 3) * ldb + pn, 0, 0);
        }
#if HAVE_TDM
        // TDM retires in order per wave: <=1 outstanding means chunk kc has landed
        if (wave == 0) {
            if (kc + 1 < nK) __builtin_amdgcn_s_wait_tensorcnt(1);
            else             __builtin_amdgcn_s_wait_tensorcnt(0);
        }
#endif
        __syncthreads();
        // A fragment: lane (M=l16, half=hlf): e<8 -> K=hlf*8+e ; e>=8 -> K=16+hlf*8+(e-8)
        union { int4 i[2]; v16bf v; } fa;
        fa.i[0] = *(const int4*)&As[bufA][wm * 16 + l16][hlf * 4];
        fa.i[1] = *(const int4*)&As[bufA][wm * 16 + l16][8 + hlf * 4];
#pragma unroll
        for (int j = 0; j < 4; ++j) {
            // B fragment: lane (N=l16, half=hlf): e<8 -> K=hlf*8+e ; e>=8 -> K=16+hlf*8+(e-8)
            const int n0 = wg * 64 + j * 16;
            union { int4 i[2]; v16bf v; } fb;
            const unsigned short* bp = &Bs[bufB][n0 + l16][hlf * 8];
            fb.i[0] = *(const int4*)(bp);
            fb.i[1] = *(const int4*)(bp + 16);
            acc[j] = __builtin_amdgcn_wmma_f32_16x16x32_bf16(
                false, fa.v, false, fb.v, (short)0, acc[j], false, false);
        }
    }

    // epilogue: C layout -> VGPR r: M = r + 8*half, N = l16
#pragma unroll
    for (int j = 0; j < 4; ++j) {
        const int n = n0blk + wg * 64 + j * 16 + l16;
        if (n >= N) continue;
        if (MODE == 0) {
            const float bb = bias0[n];
#pragma unroll
            for (int r = 0; r < 8; ++r) {
                const int mm = wm * 16 + 8 * hlf + r;
                outBF[(size_t)mm * ldo + n] = f2bf(tanhf(acc[j][r] + bb));
            }
        } else {
            const float bb = bias0[n] + bias1[n];
#pragma unroll
            for (int r = 0; r < 8; ++r) {
                const int mm = wm * 16 + 8 * hlf + r;
                const size_t o = (size_t)mm * ldo + n;
                outF[o] = acc[j][r] + bb + outF[o];
            }
        }
    }
}

// -------------------- row softmax over 50000 --------------------
__global__ __launch_bounds__(512)
void softmax_rows(float* __restrict__ x, int n) {
    __shared__ float red[512];
    const int b = blockIdx.x, t = threadIdx.x;
    float* row = x + (size_t)b * n;
    float m = -3.4e38f;
    for (int i = t; i < n; i += 512) m = fmaxf(m, row[i]);
    red[t] = m; __syncthreads();
    for (int s = 256; s > 0; s >>= 1) { if (t < s) red[t] = fmaxf(red[t], red[t + s]); __syncthreads(); }
    const float M = red[0]; __syncthreads();
    float sum = 0.f;
    for (int i = t; i < n; i += 512) { float e = __expf(row[i] - M); row[i] = e; sum += e; }
    red[t] = sum; __syncthreads();
    for (int s = 256; s > 0; s >>= 1) { if (t < s) red[t] += red[t + s]; __syncthreads(); }
    const float inv = 1.0f / red[0];
    for (int i = t; i < n; i += 512) row[i] *= inv;
}

extern "C" void kernel_launch(void* const* d_in, const int* in_sizes, int n_in,
                              void* d_out, int out_size, void* d_ws, size_t ws_size,
                              hipStream_t stream) {
    const float* enc  = (const float*)d_in[0];
    const int*   pseq = (const int*)  d_in[1];
    const int*   src  = (const int*)  d_in[2];
    const int*   lens = (const int*)  d_in[3];
    const float* Et   = (const float*)d_in[4];
    const float* W1   = (const float*)d_in[5];
    const float* b1   = (const float*)d_in[6];
    const float* W2   = (const float*)d_in[7];
    const float* b2   = (const float*)d_in[8];
    const float* W3   = (const float*)d_in[9];
    const float* b3   = (const float*)d_in[10];
    float* out = (float*)d_out;

    unsigned short* ctx  = (unsigned short*)d_ws;                                      // 64 x 2560 bf16
    unsigned short* hcat = (unsigned short*)((char*)d_ws + (size_t)B_ * CS_ * E_ * 2); // 64 x 4096 bf16

    const int tot = B_ * O_;
    zero_out_k<<<(tot + 511) / 512, 512, 0, stream>>>(out, tot);
    feat_kernel<<<B_, 512, 0, stream>>>(pseq, src, lens, out);
    gather_ctx_k<<<(B_ * CS_ * E_ + 255) / 256, 256, 0, stream>>>(pseq, Et, ctx);
    enc_to_hcat_k<<<(B_ * H_ + 255) / 256, 256, 0, stream>>>(enc, hcat);

    // h = tanh(ctx @ W1 + b1) -> hcat[:, :2048] as bf16
    gemm_wmma<0><<<(H_ + 255) / 256, 512, 0, stream>>>(
        ctx, CS_ * E_, W1, W1, CS_ * E_, CS_ * E_, H_, H_,
        b1, b1, nullptr, hcat, 2 * H_);

    // out += hcat @ [W2;W3] + b2 + b3   (819 MB weight stream, bandwidth-bound)
    gemm_wmma<1><<<(O_ + 255) / 256, 512, 0, stream>>>(
        hcat, 2 * H_, W2, W3, H_, 2 * H_, O_, O_,
        b2, b3, out, nullptr, O_);

    softmax_rows<<<B_, 512, 0, stream>>>(out, O_);
}
